// LSTMHierarchialAttention2_25340307046414
// MI455X (gfx1250) — matrix operand order repaired and checked by reading, not compiled, not run on hardware
//
#include <hip/hip_runtime.h>
#include <cstddef>

// ---------------------------------------------------------------------------
// LSTM hierarchical attention for MI455X (gfx1250, wave32, WMMA + TDM).
// S=32 sentences, W=256 words, D=512, H=1024.
//
//  1) convert inputs/weights to f16
//  2) G1 = X(8192x512) @ Wih1^T + (bih1+bhh1)       [WMMA GEMM, TDM-staged B]
//  3) persistent cooperative LSTM1 recurrence, 8192 steps
//     (h@Whh1^T via 1/16-dense WMMA matvec; weights L2-resident)
//  4) P1 = hseq1 @ attW1 + attB1                    [WMMA GEMM, TDM-staged B]
//     s1 = tanh(P1) . attU1 ; sentvec = sum_t w_t h_t
//  5) G2 = sentvecs @ Wih2^T + (bih2+bhh2)          [WMMA GEMM]
//  6) LSTM2 recurrence (32 steps, same cooperative kernel)
//  7) P2 = h2seq @ attW2 + attB2 ; doc attention ; final linear ; softmax==1
// ---------------------------------------------------------------------------

typedef _Float16 f16;
typedef __attribute__((ext_vector_type(16))) _Float16 v16h;
typedef __attribute__((ext_vector_type(8)))  _Float16 v8h;
typedef __attribute__((ext_vector_type(8)))  float    v8f;
typedef __attribute__((ext_vector_type(4)))  unsigned int u32x4;
typedef __attribute__((ext_vector_type(8)))  unsigned int u32x8;

#define RFL(x) ((unsigned)__builtin_amdgcn_readfirstlane((int)(x)))

// ---- WMMA fragment loaders (16-bit, contiguous K order shared by A and B) -----------
// A 16x32 (MxK): lane (r = lane&15, h = lane>>4): elem e -> K = 16h + e.
//                Two 16B loads.
__device__ __forceinline__ v16h load_a_frag(const f16* __restrict__ rowPtr,
                                            int k, int half) {
  v8h lo = *reinterpret_cast<const v8h*>(rowPtr + k + 16 * half);
  v8h hi = *reinterpret_cast<const v8h*>(rowPtr + k + 16 * half + 8);
  v16h a;
#pragma unroll
  for (int i = 0; i < 8; ++i) { a[i] = lo[i]; a[i + 8] = hi[i]; }
  return a;
}

// B 32x16 (KxN): lane (n = lane&15, h = lane>>4): elem e -> K = 16h + e.
// 16 contiguous f16 per lane along K (weights stored N x K row-major).
__device__ __forceinline__ v16h load_b_frag(const f16* rowPtr, int k, int half) {
  v8h lo = *reinterpret_cast<const v8h*>(rowPtr + k + 16 * half);
  v8h hi = *reinterpret_cast<const v8h*>(rowPtr + k + 16 * half + 8);
  v16h b;
#pragma unroll
  for (int i = 0; i < 8; ++i) { b[i] = lo[i]; b[i + 8] = hi[i]; }
  return b;
}

__device__ __forceinline__ float sigf(float x) { return 1.0f / (1.0f + expf(-x)); }

// ---- TDM: stage `elems` f16 (contiguous) from global into LDS --------------
// 1-D tile: tensor_dim0 = tile_dim0 = elems (<= 16384), data_size = 2 bytes,
// no multicast / padding / iteration.  Issued by one wave; EXEC is ignored.
__device__ __forceinline__ void tdm_load_to_lds(const f16* gsrc, f16* lds_dst,
                                                unsigned elems) {
  const unsigned long long ga = (unsigned long long)(size_t)gsrc;
  const unsigned lds_off = (unsigned)(size_t)lds_dst;  // flat->LDS: addr[31:0]
  u32x4 g0;
  g0[0] = RFL(1u);                                   // count=1 (user D#)
  g0[1] = RFL(lds_off);                              // lds_addr (bytes)
  g0[2] = RFL((unsigned)(ga & 0xffffffffu));         // global_addr[31:0]
  g0[3] = RFL(((unsigned)((ga >> 32) & 0x1ffffffu)) | (2u << 30)); // [56:32]|type=2
  u32x8 g1;
  g1[0] = RFL(0x00010000u);                          // wg_mask=0, data_size=1 (2B)
  g1[1] = RFL((elems & 0xffffu) << 16);              // tensor_dim0[15:0] @ bit48
  g1[2] = RFL(((elems >> 16) & 0xffffu) | (1u << 16)); // dim0[31:16]; tensor_dim1=1
  g1[3] = RFL((elems & 0xffffu) << 16);              // tile_dim0 @ bits 127:112
  g1[4] = RFL(0u);                                   // tile_dim1=0, tile_dim2=0
  g1[5] = RFL(elems);                                // tensor_dim0_stride[31:0]
  g1[6] = RFL(0u);                                   // stride0[47:32], stride1[15:0]
  g1[7] = RFL(0u);
  asm volatile("tensor_load_to_lds %0, %1" :: "s"(g0), "s"(g1) : "memory");
  __builtin_amdgcn_s_wait_tensorcnt(0);              // TENSORcnt is per-wave
}

// ---- GEMM: C(MxN) = A(MxK) @ B(NxK)^T + bias[n] ---------------------------
// grid.x = N/16 tiles, grid.y = ceil(M/128); 256 threads = 8 waves.
// All 8 waves share tileN, so the 16xK B-slice is staged ONCE per block into
// LDS by the Tensor Data Mover (8x less L2 traffic than per-wave loads).
__global__ __launch_bounds__(256) void wmma_gemm_nt(
    const f16* __restrict__ A, const f16* __restrict__ B,
    const float* __restrict__ bias, float* __restrict__ C,
    int M, int N, int K) {
  __shared__ f16 Bs[16 * 1024];                 // 32KB, supports K <= 1024
  const int lane  = threadIdx.x & 31;
  const int wave  = threadIdx.x >> 5;
  const int tileN = blockIdx.x << 4;
  const int tileM = ((blockIdx.y << 3) + wave) << 4;
  const bool active = (tileM < M);

  if (wave == 0)                                // one TDM issue per block
    tdm_load_to_lds(B + (size_t)tileN * K, Bs, (unsigned)(16 * K));
  __syncthreads();                              // publish LDS tile
  if (!active) return;

  const int r    = lane & 15;
  const int half = lane >> 4;
  const f16* Arow = A + (size_t)(tileM + r) * K;
  const f16* Brow = Bs + r * K;                 // column n = lane&15, from LDS
  v8f acc = {};
  for (int k = 0; k < K; k += 32) {
    __builtin_prefetch(Arow + k + 256, 0, 1);   // global_prefetch_b8
    v16h a = load_a_frag(Arow, k, half);
    v16h b = load_b_frag(Brow, k, half);        // ds_load_b128 x2
    acc = __builtin_amdgcn_wmma_f32_16x16x32_f16(
        false, a, false, b, (short)0, acc, false, false);
  }
  // D layout: lanes 0-15 col=lane rows M=0..7; lanes 16-31 rows M=8..15
  const int   col     = tileN + r;
  const int   rowBase = tileM + (half << 3);
  const float bv      = bias ? bias[col] : 0.0f;
#pragma unroll
  for (int v = 0; v < 8; ++v)
    C[(size_t)(rowBase + v) * N + col] = acc[v] + bv;
}

// ---- Persistent cooperative LSTM recurrence --------------------------------
// grid = 32 WGs x 256 thr. WG owns 32 hidden units (jBase..+32): its 4x32
// gate rows of Whh (4H x H, f16, row-major). Wave w computes 16 gate rows via
// WMMA (h broadcast as B column 0). Grid barrier per step via atomic counter.
__global__ __launch_bounds__(256) void lstm_recurrent(
    const float* __restrict__ Gpre,   // (T, 4H) x-part + biases
    const f16*   __restrict__ Whh,    // (4H, H) f16
    float* __restrict__ hseq,         // (T, H) f32 out
    f16*   __restrict__ hseq_h,       // (T, H) f16 out
    float* hbuf, f16* hbuf_h, float* cbuf,
    unsigned int* sync, int T, int Hdim) {
  const int H4    = Hdim << 2;
  const int tid   = threadIdx.x;
  const int lane  = tid & 31;
  const int wave  = tid >> 5;
  const int Hc    = Hdim / gridDim.x;        // 32
  const int jBase = blockIdx.x * Hc;
  const int gate  = wave >> 1;               // i,f,g,o
  const int sub   = (wave & 1) << 4;         // 0 or 16 within the 32-row gate slice
  const int r     = lane & 15;
  const int half  = lane >> 4;
  const int gRow0 = gate * Hdim + jBase + sub;
  const f16* Arow = Whh + (size_t)(gRow0 + r) * Hdim;
  __shared__ float gsh[128];                 // 4 gates x 32 units

  for (int t = 0; t < T; ++t) {
    v8f acc = {};
    for (int k = 0; k < Hdim; k += 32) {
      v16h a = load_a_frag(Arow, k, half);
      v16h b = {};
      if (r == 0) {                          // only N-column 0 carries h
        v8h lo = *reinterpret_cast<const v8h*>(hbuf_h + k + 16 * half);
        v8h hi = *reinterpret_cast<const v8h*>(hbuf_h + k + 16 * half + 8);
#pragma unroll
        for (int i = 0; i < 8; ++i) { b[i] = lo[i]; b[i + 8] = hi[i]; }
      }
      acc = __builtin_amdgcn_wmma_f32_16x16x32_f16(
          false, a, false, b, (short)0, acc, false, false);
    }
    if (r == 0) {                            // lanes 0/16 hold D column 0
      const int base = sub + (half << 3);
#pragma unroll
      for (int v = 0; v < 8; ++v) {
        const int grow = gRow0 + (half << 3) + v;
        gsh[(gate << 5) + base + v] = acc[v] + Gpre[(size_t)t * H4 + grow];
      }
    }
    __syncthreads();
    if (tid < Hc) {
      const int   j  = jBase + tid;
      const float iv = gsh[tid], fv = gsh[32 + tid];
      const float gv = gsh[64 + tid], ov = gsh[96 + tid];
      const float cv = sigf(fv) * cbuf[j] + sigf(iv) * tanhf(gv);
      const float hv = sigf(ov) * tanhf(cv);
      cbuf[j] = cv;
      hbuf[j] = hv;
      hbuf_h[j] = (f16)hv;
      hseq[(size_t)t * Hdim + j]   = hv;
      hseq_h[(size_t)t * Hdim + j] = (f16)hv;
    }
    __threadfence();
    __syncthreads();
    if (tid == 0) {                          // grid barrier: arrive + spin
      __hip_atomic_fetch_add(sync, 1u, __ATOMIC_RELEASE, __HIP_MEMORY_SCOPE_AGENT);
      const unsigned target = gridDim.x * (unsigned)(t + 1);
      while (__hip_atomic_load(sync, __ATOMIC_ACQUIRE, __HIP_MEMORY_SCOPE_AGENT) < target)
        __builtin_amdgcn_s_sleep(1);
    }
    __syncthreads();
  }
}

// ---- small helper kernels --------------------------------------------------
__global__ void f32_to_f16_k(const float* __restrict__ in, f16* __restrict__ out, int n) {
  int i = blockIdx.x * 256 + threadIdx.x;
  if (i < n) out[i] = (f16)in[i];
}

// out (N x K) f16 = transpose of in (K x N) f32
__global__ void transpose_to_f16(const float* __restrict__ in, f16* __restrict__ out,
                                 int Kd, int Nd) {
  int idx = blockIdx.x * 256 + threadIdx.x;
  if (idx >= Kd * Nd) return;
  int n = idx / Kd, k = idx % Kd;
  out[idx] = (f16)in[(size_t)k * Nd + n];
}

__global__ void add_vec_k(const float* __restrict__ a, const float* __restrict__ b,
                          float* __restrict__ o, int n) {
  int i = blockIdx.x * 256 + threadIdx.x;
  if (i < n) o[i] = a[i] + b[i];
}

__global__ void init_state_k(const float* __restrict__ h0, const float* __restrict__ c0,
                             float* hbuf, f16* hbuf_h, float* cbuf,
                             unsigned int* sync, int Hdim) {
  int i = blockIdx.x * 256 + threadIdx.x;
  if (i == 0) *sync = 0u;
  if (i < Hdim) { hbuf[i] = h0[i]; hbuf_h[i] = (f16)h0[i]; cbuf[i] = c0[i]; }
}

// s[m] = sum_n tanh(P[m,n]) * U[n]   (one WG per row, LDS tree reduce)
__global__ __launch_bounds__(256) void attn_score_k(const float* __restrict__ P,
                                                    const float* __restrict__ U,
                                                    float* __restrict__ s, int Hdim) {
  __shared__ float red[256];
  const int m = blockIdx.x;
  float acc = 0.0f;
  for (int n = threadIdx.x; n < Hdim; n += 256)
    acc += tanhf(P[(size_t)m * Hdim + n]) * U[n];
  red[threadIdx.x] = acc;
  __syncthreads();
  for (int off = 128; off > 0; off >>= 1) {
    if (threadIdx.x < off) red[threadIdx.x] += red[threadIdx.x + off];
    __syncthreads();
  }
  if (threadIdx.x == 0) s[m] = red[0];
}

// out[sent,:] = sum_t w_t * hseq[sent,t,:], w_t = e^s/(e^s + e)
__global__ __launch_bounds__(256) void attn_combine_k(const float* __restrict__ hseq,
                                                      const float* __restrict__ s,
                                                      float* __restrict__ outv,
                                                      int T, int Hdim) {
  __shared__ float w[256];
  const int sent = blockIdx.x;
  const float E = 2.718281828459045f;
  for (int t = threadIdx.x; t < T; t += 256) {
    float es = expf(s[sent * T + t]);
    w[t] = es / (es + E);
  }
  __syncthreads();
  const float* hs = hseq + (size_t)sent * T * Hdim;
  for (int n = threadIdx.x; n < Hdim; n += 256) {
    float a = 0.0f;
    for (int t = 0; t < T; ++t) a += hs[(size_t)t * Hdim + n] * w[t];
    outv[(size_t)sent * Hdim + n] = a;
  }
}

__global__ __launch_bounds__(256) void final_softmax_k(const float* __restrict__ doc,
                                                       const float* __restrict__ Wf,
                                                       const float* __restrict__ bf_,
                                                       float* __restrict__ out, int Hdim) {
  __shared__ float red[256];
  float acc = 0.0f;
  for (int n = threadIdx.x; n < Hdim; n += 256) acc += doc[n] * Wf[n];
  red[threadIdx.x] = acc;
  __syncthreads();
  for (int off = 128; off > 0; off >>= 1) {
    if (threadIdx.x < off) red[threadIdx.x] += red[threadIdx.x + off];
    __syncthreads();
  }
  if (threadIdx.x == 0) {
    float logit = red[0] + bf_[0];
    out[0] = expf(logit - logit);   // softmax over one class == 1.0
  }
}

// ---------------------------------------------------------------------------
extern "C" void kernel_launch(void* const* d_in, const int* in_sizes, int n_in,
                              void* d_out, int out_size, void* d_ws, size_t ws_size,
                              hipStream_t stream) {
  (void)in_sizes; (void)n_in; (void)out_size; (void)ws_size;
  constexpr int S = 32, W = 256, D = 512, H = 1024, H4 = 4096, SW = S * W;

  const float* inputs = (const float*)d_in[0];
  const float* h1_0   = (const float*)d_in[1];
  const float* c1_0   = (const float*)d_in[2];
  const float* h2_0   = (const float*)d_in[3];
  const float* c2_0   = (const float*)d_in[4];
  const float* Wih1   = (const float*)d_in[5];
  const float* Whh1   = (const float*)d_in[6];
  const float* bih1   = (const float*)d_in[7];
  const float* bhh1   = (const float*)d_in[8];
  const float* Wih2   = (const float*)d_in[9];
  const float* Whh2   = (const float*)d_in[10];
  const float* bih2   = (const float*)d_in[11];
  const float* bhh2   = (const float*)d_in[12];
  const float* attW1  = (const float*)d_in[13];
  const float* attB1  = (const float*)d_in[14];
  const float* attU1  = (const float*)d_in[15];
  const float* attW2  = (const float*)d_in[16];
  const float* attB2  = (const float*)d_in[17];
  const float* attU2  = (const float*)d_in[18];
  const float* Wf     = (const float*)d_in[19];
  const float* bf_    = (const float*)d_in[20];
  float* out = (float*)d_out;

  // workspace carve-up (256B aligned regions)
  size_t off = 0;
  auto take = [&](size_t bytes) -> void* {
    void* p = (void*)((char*)d_ws + off);
    off += (bytes + 255) & ~((size_t)255);
    return p;
  };
  f16*   Xh     = (f16*)  take((size_t)SW * D * 2);
  f16*   Wih1h  = (f16*)  take((size_t)H4 * D * 2);
  f16*   Whh1h  = (f16*)  take((size_t)H4 * H * 2);
  f16*   Wih2h  = (f16*)  take((size_t)H4 * H * 2);
  f16*   Whh2h  = (f16*)  take((size_t)H4 * H * 2);
  f16*   aW1T   = (f16*)  take((size_t)H * H * 2);
  f16*   aW2T   = (f16*)  take((size_t)H * H * 2);
  float* b1     = (float*)take((size_t)H4 * 4);
  float* b2     = (float*)take((size_t)H4 * 4);
  float* G1     = (float*)take((size_t)SW * H4 * 4);
  float* hseq1  = (float*)take((size_t)SW * H * 4);
  f16*   hseq1h = (f16*)  take((size_t)SW * H * 2);
  float* P1     = (float*)take((size_t)SW * H * 4);
  float* s1     = (float*)take((size_t)SW * 4);
  float* sent   = (float*)take((size_t)S * H * 4);
  f16*   senth  = (f16*)  take((size_t)S * H * 2);
  float* G2     = (float*)take((size_t)S * H4 * 4);
  float* hseq2  = (float*)take((size_t)S * H * 4);
  f16*   hseq2h = (f16*)  take((size_t)S * H * 2);
  float* P2     = (float*)take((size_t)S * H * 4);
  float* s2     = (float*)take((size_t)S * 4);
  float* doc    = (float*)take((size_t)H * 4);
  float* hb1    = (float*)take((size_t)H * 4);
  f16*   hb1h   = (f16*)  take((size_t)H * 2);
  float* cb1    = (float*)take((size_t)H * 4);
  float* hb2    = (float*)take((size_t)H * 4);
  f16*   hb2h   = (f16*)  take((size_t)H * 2);
  float* cb2    = (float*)take((size_t)H * 4);
  unsigned* sy1 = (unsigned*)take(256);
  unsigned* sy2 = (unsigned*)take(256);

  auto cdiv = [](int a, int b) { return (a + b - 1) / b; };

  // 1) precision conversions + fused biases + transposed attention weights
  f32_to_f16_k<<<cdiv(SW * D, 256), 256, 0, stream>>>(inputs, Xh, SW * D);
  f32_to_f16_k<<<cdiv(H4 * D, 256), 256, 0, stream>>>(Wih1, Wih1h, H4 * D);
  f32_to_f16_k<<<cdiv(H4 * H, 256), 256, 0, stream>>>(Whh1, Whh1h, H4 * H);
  f32_to_f16_k<<<cdiv(H4 * H, 256), 256, 0, stream>>>(Wih2, Wih2h, H4 * H);
  f32_to_f16_k<<<cdiv(H4 * H, 256), 256, 0, stream>>>(Whh2, Whh2h, H4 * H);
  transpose_to_f16<<<cdiv(H * H, 256), 256, 0, stream>>>(attW1, aW1T, H, H);
  transpose_to_f16<<<cdiv(H * H, 256), 256, 0, stream>>>(attW2, aW2T, H, H);
  add_vec_k<<<cdiv(H4, 256), 256, 0, stream>>>(bih1, bhh1, b1, H4);
  add_vec_k<<<cdiv(H4, 256), 256, 0, stream>>>(bih2, bhh2, b2, H4);

  // 2) G1 = X @ Wih1^T + b1   (8192x512 x 512x4096)
  wmma_gemm_nt<<<dim3(H4 / 16, cdiv(SW, 128)), 256, 0, stream>>>(
      Xh, Wih1h, b1, G1, SW, H4, D);

  // 3) LSTM1 recurrence over all 8192 steps (state carries across sentences)
  init_state_k<<<cdiv(H, 256), 256, 0, stream>>>(h1_0, c1_0, hb1, hb1h, cb1, sy1, H);
  lstm_recurrent<<<dim3(32), 256, 0, stream>>>(G1, Whh1h, hseq1, hseq1h,
                                               hb1, hb1h, cb1, sy1, SW, H);

  // 4) sentence attention: P1 = hseq1 @ attW1 + attB1 ; s1 ; sentvecs
  wmma_gemm_nt<<<dim3(H / 16, cdiv(SW, 128)), 256, 0, stream>>>(
      hseq1h, aW1T, attB1, P1, SW, H, H);
  attn_score_k<<<SW, 256, 0, stream>>>(P1, attU1, s1, H);
  attn_combine_k<<<S, 256, 0, stream>>>(hseq1, s1, sent, W, H);
  f32_to_f16_k<<<cdiv(S * H, 256), 256, 0, stream>>>(sent, senth, S * H);

  // 5) G2 = sentvecs @ Wih2^T + b2   (32x1024 x 1024x4096)
  wmma_gemm_nt<<<dim3(H4 / 16, cdiv(S, 128)), 256, 0, stream>>>(
      senth, Wih2h, b2, G2, S, H4, H);

  // 6) LSTM2 recurrence (32 steps)
  init_state_k<<<cdiv(H, 256), 256, 0, stream>>>(h2_0, c2_0, hb2, hb2h, cb2, sy2, H);
  lstm_recurrent<<<dim3(32), 256, 0, stream>>>(G2, Whh2h, hseq2, hseq2h,
                                               hb2, hb2h, cb2, sy2, S, H);

  // 7) document attention + final linear + softmax(1x1)
  wmma_gemm_nt<<<dim3(H / 16, cdiv(S, 128)), 256, 0, stream>>>(
      hseq2h, aW2T, attB2, P2, S, H, H);
  attn_score_k<<<S, 256, 0, stream>>>(P2, attU2, s2, H);
  attn_combine_k<<<1, 256, 0, stream>>>(hseq2, s2, doc, S, H);
  final_softmax_k<<<1, 256, 0, stream>>>(doc, Wf, bf_, out, H);
}
